// H2XAttention_66864050864778
// MI455X (gfx1250) — matrix operand order repaired and checked by reading, not compiled, not run on hardware
//
#include <hip/hip_runtime.h>
#include <hip/hip_bf16.h>
#include <math.h>

#define N_NODES 10000
#define E_EDGES 320000
#define NH      16
#define KV      340
#define KVP     352            // padded to 11*32
#define LN_EPS  1e-5f

typedef __attribute__((ext_vector_type(16))) __bf16 v16bf;
typedef __attribute__((ext_vector_type(8)))  __bf16 v8bf;
typedef __attribute__((ext_vector_type(8)))  float  v8f;

// ---- packed bf16 weight regions (element offsets) ------------------------
// Each "block" = one 16x16x32 WMMA B-operand tile: 32 lanes x 16 bf16 = 512.
#define L1_NT 16              // layer1 combined [352 x 256]: 11 ktiles x 16 ntiles
#define L1_KT 11
#define L2_NT 9               // layer2 combined [128 x 144]: 4 x 9
#define L2_KT 4
#define Q_NT  8               // q layers [128 x 128]: 4 x 8
#define Q_KT  4
#define L1_OFF 0
#define L2_OFF (L1_OFF + L1_KT*L1_NT*512)
#define Q1_OFF (L2_OFF + L2_KT*L2_NT*512)
#define Q2_OFF (Q1_OFF + Q_KT*Q_NT*512)
#define PK_TOTAL (Q2_OFF + Q_KT*Q_NT*512)

// ---- WMMA helpers --------------------------------------------------------
__device__ __forceinline__ v8f wmma_bf16(v16bf a, v16bf b, v8f c) {
  return __builtin_amdgcn_wmma_f32_16x16x32_bf16(false, a, false, b, (short)0, c,
                                                 false, false);
}

// A operand (16x32 bf16 row tile) from LDS. ISA layout: lanes 0-15 hold
// K = {0..7, 16..23}, lanes 16-31 hold K = {8..15, 24..31} of row M=lane%16.
__device__ __forceinline__ v16bf load_a(const __bf16* lds, int stride, int k0,
                                        unsigned lane) {
  unsigned lrow  = lane & 15u;
  unsigned khalf = (lane >> 4) << 3;        // 0 or 8
  const __bf16* p = lds + (int)lrow * stride + k0 + (int)khalf;
  v8bf lo = *(const v8bf*)(p);
  v8bf hi = *(const v8bf*)(p + 16);
  v16bf a;
#pragma unroll
  for (int i = 0; i < 8; ++i) { a[i] = lo[i]; a[i + 8] = hi[i]; }
  return a;
}

// B operand: pre-swizzled contiguous 32B per lane.
__device__ __forceinline__ v16bf load_b(const __bf16* pk, int block, unsigned lane) {
  return *(const v16bf*)(pk + (size_t)block * 512 + (size_t)lane * 16);
}

// C (16x16 f32): VGPR r, lanes 0-15 -> M=r, lanes 16-31 -> M=r+8; N=lane%16.
__device__ __forceinline__ void store_c(float* dst, int stride, int n0, v8f c,
                                        unsigned lane) {
  unsigned lrow = lane & 15u;
  unsigned crow = (lane >> 4) << 3;
#pragma unroll
  for (int r = 0; r < 8; ++r)
    dst[(int)(crow + r) * stride + n0 + (int)lrow] = c[r];
}

__device__ __forceinline__ void atomicMaxF(float* addr, float v) {
  if (v >= 0.f) atomicMax((int*)addr, __float_as_int(v));
  else          atomicMin((unsigned int*)addr, __float_as_uint(v));
}

// ---- weight packing ------------------------------------------------------
__global__ void k_pack(const float* kW1, const float* vW1, const float* kW2,
                       const float* vW2, const float* qW1, const float* qW2,
                       __bf16* pk) {
  int i = blockIdx.x * 256 + threadIdx.x;
  if (i >= PK_TOTAL) return;
  float val = 0.f;
  int idx, NT;
  const enum { RL1, RL2, RQ1, RQ2 } reg =
      (i < L2_OFF) ? RL1 : (i < Q1_OFF) ? RL2 : (i < Q2_OFF) ? RQ1 : RQ2;
  idx = i - ((reg == RL1) ? L1_OFF : (reg == RL2) ? L2_OFF
             : (reg == RQ1) ? Q1_OFF : Q2_OFF);
  NT  = (reg == RL1) ? L1_NT : (reg == RL2) ? L2_NT : Q_NT;
  int blk  = idx >> 9, r = idx & 511;
  int lane = r >> 4,  elem = r & 15;
  int kt = blk / NT, nt = blk % NT;
  int k = kt * 32 + ((lane >> 4) << 3) + (elem & 7) + ((elem >> 3) << 4);
  int n = nt * 16 + (lane & 15);
  if (reg == RL1) {                         // [352(pad) x 256] = [xk_W1 | xv_W1]
    if (k < KV) val = (n < 128) ? kW1[k * 128 + n] : vW1[k * 128 + (n - 128)];
  } else if (reg == RL2) {                  // [128 x 144] = [xk_W2 | xv_W2]
    val = (n < 128) ? kW2[k * 128 + n] : vW2[k * 16 + (n - 128)];
  } else if (reg == RQ1) {
    val = qW1[k * 128 + n];
  } else {
    val = qW2[k * 128 + n];
  }
  pk[i] = (__bf16)val;
}

// ---- init scratch --------------------------------------------------------
__global__ void k_init(float* smax, float* denom, float* out) {
  int i = blockIdx.x * 256 + threadIdx.x;
  if (i < N_NODES * NH) { smax[i] = -3.0e38f; denom[i] = 0.f; }
  if (i < N_NODES * 3)  out[i] = 0.f;
}

// ---- q MLP: h[N,128] -> q[N,128] ----------------------------------------
__global__ __launch_bounds__(128)
void k_qmlp(const float* __restrict__ h, const float* __restrict__ b1,
            const float* __restrict__ g, const float* __restrict__ bt,
            const float* __restrict__ b2, const __bf16* __restrict__ pk,
            float* __restrict__ qout) {
  __shared__ __attribute__((aligned(32))) __bf16 s_in[16 * 128];
  __shared__ __attribute__((aligned(32))) float  s_h1[16 * 128];
  __shared__ __attribute__((aligned(32))) __bf16 s_hb[16 * 128];
  int t = threadIdx.x;
  unsigned lane = t & 31u;
  int w = t >> 5;
  int nbase = blockIdx.x * 16;

  for (int i = t; i < 16 * 128; i += 128) {
    int rl = i >> 7, col = i & 127;
    int node = nbase + rl;
    s_in[i] = (__bf16)((node < N_NODES) ? h[node * 128 + col] : 0.f);
  }
  __syncthreads();

  const __bf16* pk1 = pk + Q1_OFF;
  for (int nt = w; nt < Q_NT; nt += 4) {
    v8f c = {};
#pragma unroll
    for (int kt = 0; kt < Q_KT; ++kt)
      c = wmma_bf16(load_a(s_in, 128, kt * 32, lane),
                    load_b(pk1, kt * Q_NT + nt, lane), c);
    store_c(s_h1, 128, nt * 16, c, lane);
  }
  __syncthreads();

  if (t < 16) {
    float mu = 0.f, ss = 0.f;
    for (int j = 0; j < 128; ++j) { float y = s_h1[t*128+j] + b1[j]; mu += y; ss += y*y; }
    mu *= (1.f / 128.f);
    float rs = rsqrtf(ss * (1.f / 128.f) - mu * mu + LN_EPS);
    for (int j = 0; j < 128; ++j) {
      float y = s_h1[t*128+j] + b1[j];
      float z = (y - mu) * rs * g[j] + bt[j];
      s_hb[t*128+j] = (__bf16)(z > 0.f ? z : 0.f);
    }
  }
  __syncthreads();

  const __bf16* pk2 = pk + Q2_OFF;
  for (int nt = w; nt < Q_NT; nt += 4) {
    v8f c = {};
#pragma unroll
    for (int kt = 0; kt < Q_KT; ++kt)
      c = wmma_bf16(load_a(s_hb, 128, kt * 32, lane),
                    load_b(pk2, kt * Q_NT + nt, lane), c);
    unsigned lrow = lane & 15u, crow = (lane >> 4) << 3;
#pragma unroll
    for (int r = 0; r < 8; ++r) {
      int node = nbase + (int)crow + r;
      if (node < N_NODES) {
        int col = nt * 16 + (int)lrow;
        qout[node * 128 + col] = c[r] + b2[col];
      }
    }
  }
}

// ---- edge kernel: features + k/v MLPs + logits + seg-max -----------------
__global__ __launch_bounds__(128)
void k_edge(const float* __restrict__ x, const float* __restrict__ h,
            const float* __restrict__ ea, const int* __restrict__ eidx,
            const float* kb1, const float* kg, const float* kbt, const float* kb2,
            const float* vb1, const float* vg, const float* vbt, const float* vb2,
            const __bf16* __restrict__ pk, const float* __restrict__ qbuf,
            float* __restrict__ logits, float* __restrict__ vout,
            float* __restrict__ relx_out, float* __restrict__ smax) {
  __shared__ __attribute__((aligned(32))) __bf16 s_feat[16 * KVP];  // 11 KB
  __shared__ __attribute__((aligned(32))) float  s_h1[16 * 256];    // 16 KB
  __shared__ __attribute__((aligned(32))) __bf16 s_hb[16 * 256];    // 8 KB
  __shared__ __attribute__((aligned(32))) float  s_o2[16 * 144];    // 9 KB
  __shared__ int   s_src[16], s_dst[16];
  __shared__ float s_rx[16][3], s_dist[16];

  int t = threadIdx.x;
  unsigned lane = t & 31u;
  int w = t >> 5;
  int ebase = blockIdx.x * 16;

  if (t < 16) {
    int e = ebase + t;
    int src = 0, dst = 0;
    float r0 = 0.f, r1 = 0.f, r2 = 0.f;
    if (e < E_EDGES) {
      src = eidx[e]; dst = eidx[E_EDGES + e];
      r0 = x[dst*3+0] - x[src*3+0];
      r1 = x[dst*3+1] - x[src*3+1];
      r2 = x[dst*3+2] - x[src*3+2];
    }
    s_src[t] = src; s_dst[t] = dst;
    s_rx[t][0] = r0; s_rx[t][1] = r1; s_rx[t][2] = r2;
    s_dist[t] = sqrtf(r0*r0 + r1*r1 + r2*r2);
    if (e < E_EDGES) {
      relx_out[e*3+0] = r0; relx_out[e*3+1] = r1; relx_out[e*3+2] = r2;
    }
  }
  __syncthreads();

  { // build KV features: [4 attr | 80 attr⊗gauss | 128 h[dst] | 128 h[src] | pad]
    int el = t >> 3, c0 = t & 7;
    int e = ebase + el;
    bool ok = (e < E_EDGES);
    int src = s_src[el], dst = s_dst[el];
    float dist = s_dist[el];
    const float step  = 10.0f / 19.0f;
    const float coeff = -0.5f / (step * step);
    for (int c = c0; c < KVP; c += 8) {
      float v = 0.f;
      if (ok) {
        if (c < 4) v = ea[e*4 + c];
        else if (c < 84) {
          int j = c - 4, ef = j / 20, ng = j % 20;
          float d = dist - step * (float)ng;
          v = ea[e*4 + ef] * __expf(coeff * d * d);
        } else if (c < 212) v = h[dst*128 + (c - 84)];
        else if (c < 340)   v = h[src*128 + (c - 212)];
      }
      s_feat[el * KVP + c] = (__bf16)v;
    }
  }
  __syncthreads();

  // layer 1: [16 x 352] @ [352 x 256]  (k-hidden | v-hidden fused)
  const __bf16* pk1 = pk + L1_OFF;
  for (int nt = w * 4; nt < w * 4 + 4; ++nt) {
    v8f c = {};
#pragma unroll
    for (int kt = 0; kt < L1_KT; ++kt)
      c = wmma_bf16(load_a(s_feat, KVP, kt * 32, lane),
                    load_b(pk1, kt * L1_NT + nt, lane), c);
    store_c(s_h1, 256, nt * 16, c, lane);
  }
  __syncthreads();

  // bias + LayerNorm + ReLU per (edge, half)
  if (t < 32) {
    int el = t >> 1, half = t & 1;
    const float* b1 = half ? vb1 : kb1;
    const float* gg = half ? vg  : kg;
    const float* bb = half ? vbt : kbt;
    int base = el * 256 + half * 128;
    float mu = 0.f, ss = 0.f;
    for (int j = 0; j < 128; ++j) { float y = s_h1[base+j] + b1[j]; mu += y; ss += y*y; }
    mu *= (1.f / 128.f);
    float rs = rsqrtf(ss * (1.f / 128.f) - mu * mu + LN_EPS);
    for (int j = 0; j < 128; ++j) {
      float y = s_h1[base+j] + b1[j];
      float z = (y - mu) * rs * gg[j] + bb[j];
      s_hb[base+j] = (__bf16)(z > 0.f ? z : 0.f);
    }
  }
  __syncthreads();

  // layer 2: k: [16x128]@[128x128] (nt 0..7, A=half0); v: [16x128]@[128x16] (nt 8, A=half1)
  const __bf16* pk2 = pk + L2_OFF;
  for (int nt = w; nt < L2_NT; nt += 4) {
    const __bf16* abase = s_hb + ((nt < 8) ? 0 : 128);
    v8f c = {};
#pragma unroll
    for (int kt = 0; kt < L2_KT; ++kt)
      c = wmma_bf16(load_a(abase, 256, kt * 32, lane),
                    load_b(pk2, kt * L2_NT + nt, lane), c);
    store_c(s_o2, 144, nt * 16, c, lane);
  }
  __syncthreads();

  // v output
  if (t < 16) {
    int e = ebase + t;
    if (e < E_EDGES) {
#pragma unroll
      for (int hh = 0; hh < NH; ++hh)
        vout[e * NH + hh] = s_o2[t * 144 + 128 + hh] + vb2[hh];
    }
  }
  // logits = (q[dst] * k).sum(head) / sqrt(8)  + seg-max atomic
  for (int task = t; task < 16 * NH; task += 128) {
    int el = task >> 4, hh = task & 15;
    int e = ebase + el;
    if (e < E_EDGES) {
      int d = s_dst[el];
      float acc = 0.f;
#pragma unroll
      for (int dd = 0; dd < 8; ++dd) {
        float kk = s_o2[el * 144 + hh * 8 + dd] + kb2[hh * 8 + dd];
        acc += kk * qbuf[d * 128 + hh * 8 + dd];
      }
      float lg = acc * 0.35355339059327373f;   // 1/sqrt(8)
      logits[e * NH + hh] = lg;
      atomicMaxF(&smax[d * NH + hh], lg);
    }
  }
}

// ---- softmax pass 2: exp + denominator ----------------------------------
__global__ void k_expsum(const int* __restrict__ eidx, const float* __restrict__ smax,
                         float* __restrict__ logits, float* __restrict__ denom) {
  int i = blockIdx.x * 256 + threadIdx.x;
  if (i >= E_EDGES * NH) return;
  int e = i >> 4, hh = i & 15;
  int d = eidx[E_EDGES + e];
  float ex = __expf(logits[i] - smax[d * NH + hh]);
  logits[i] = ex;
  atomicAdd(&denom[d * NH + hh], ex);
}

// ---- final: alpha * v * rel_x, scatter-sum, head mean -------------------
__global__ void k_final(const int* __restrict__ eidx, const float* __restrict__ ex,
                        const float* __restrict__ denom, const float* __restrict__ vbuf,
                        const float* __restrict__ relx, float* __restrict__ out) {
  int e = blockIdx.x * 256 + threadIdx.x;
  if (e >= E_EDGES) return;
  int d = eidx[E_EDGES + e];
  float s = 0.f;
#pragma unroll
  for (int hh = 0; hh < NH; ++hh)
    s += (ex[e * NH + hh] / denom[d * NH + hh]) * vbuf[e * NH + hh];
  s *= (1.f / (float)NH);
  atomicAdd(&out[d * 3 + 0], s * relx[e * 3 + 0]);
  atomicAdd(&out[d * 3 + 1], s * relx[e * 3 + 1]);
  atomicAdd(&out[d * 3 + 2], s * relx[e * 3 + 2]);
}

extern "C" void kernel_launch(void* const* d_in, const int* in_sizes, int n_in,
                              void* d_out, int out_size, void* d_ws, size_t ws_size,
                              hipStream_t stream) {
  const float* x    = (const float*)d_in[0];
  const float* h    = (const float*)d_in[1];
  const float* ea   = (const float*)d_in[2];
  const int*   eidx = (const int*)d_in[4];
  const float* kW1 = (const float*)d_in[5],  *kb1 = (const float*)d_in[6];
  const float* kg  = (const float*)d_in[7],  *kbt = (const float*)d_in[8];
  const float* kW2 = (const float*)d_in[9],  *kb2 = (const float*)d_in[10];
  const float* vW1 = (const float*)d_in[11], *vb1 = (const float*)d_in[12];
  const float* vg  = (const float*)d_in[13], *vbt = (const float*)d_in[14];
  const float* vW2 = (const float*)d_in[15], *vb2 = (const float*)d_in[16];
  const float* qW1 = (const float*)d_in[17], *qb1 = (const float*)d_in[18];
  const float* qg  = (const float*)d_in[19], *qbt = (const float*)d_in[20];
  const float* qW2 = (const float*)d_in[21], *qb2 = (const float*)d_in[22];

  float* qbuf   = (float*)d_ws;                       // N*128
  float* logits = qbuf + (size_t)N_NODES * 128;       // E*16 (becomes exp())
  float* vbuf   = logits + (size_t)E_EDGES * NH;      // E*16
  float* relx   = vbuf + (size_t)E_EDGES * NH;        // E*3
  float* smax   = relx + (size_t)E_EDGES * 3;         // N*16
  float* denom  = smax + (size_t)N_NODES * NH;        // N*16
  __bf16* pk    = (__bf16*)(denom + (size_t)N_NODES * NH);  // PK_TOTAL (32B-aligned)
  float* out    = (float*)d_out;

  k_init<<<(N_NODES * NH + 255) / 256, 256, 0, stream>>>(smax, denom, out);
  k_pack<<<(PK_TOTAL + 255) / 256, 256, 0, stream>>>(kW1, vW1, kW2, vW2, qW1, qW2, pk);
  k_qmlp<<<(N_NODES + 15) / 16, 128, 0, stream>>>(h, qb1, qg, qbt, qb2, pk, qbuf);
  k_edge<<<E_EDGES / 16, 128, 0, stream>>>(x, h, ea, eidx,
                                           kb1, kg, kbt, kb2,
                                           vb1, vg, vbt, vb2,
                                           pk, qbuf, logits, vbuf, relx, smax);
  k_expsum<<<(E_EDGES * NH + 255) / 256, 256, 0, stream>>>(eidx, smax, logits, denom);
  k_final<<<(E_EDGES + 255) / 256, 256, 0, stream>>>(eidx, logits, denom, vbuf, relx, out);
}
